// MultiHeadSelfAttention_24412594110765
// MI455X (gfx1250) — compile-verified, matrix operand order checked by reading, not run on hardware
//
#include <hip/hip_runtime.h>

typedef unsigned short ushort_t;
typedef __attribute__((ext_vector_type(16))) __bf16 v16bf;
typedef __attribute__((ext_vector_type(8)))  __bf16 v8bf;
typedef __attribute__((ext_vector_type(8)))  float  v8f;
typedef __attribute__((ext_vector_type(4))) unsigned int u32x4;
typedef __attribute__((ext_vector_type(8))) int i32x8;
typedef __attribute__((ext_vector_type(4))) int i32x4;

#define B_  4
#define S_  2048
#define D_  1024
#define H_  16
#define DK_ 64

#if __has_builtin(__builtin_amdgcn_tensor_load_to_lds) && __has_builtin(__builtin_amdgcn_s_wait_tensorcnt)
#define HAVE_TDM 1
#else
#define HAVE_TDM 0
#endif

// ---------------- helpers ----------------

__device__ __forceinline__ ushort_t f2bf(float f) {
    unsigned int u = __float_as_uint(f);
    u += 0x7FFFu + ((u >> 16) & 1u);   // round-to-nearest-even
    return (ushort_t)(u >> 16);
}

__device__ __forceinline__ v8f vzero8() {
    v8f z = {0.f, 0.f, 0.f, 0.f, 0.f, 0.f, 0.f, 0.f};
    return z;
}

// Load a 16x32 bf16 fragment in the CDNA5 WMMA A/B register layout from a
// row-major (stride in elements) matrix. Per ISA 7.12.2 (16-bit 16x32):
//   lanes 0-15 : row = lane,    K = {0..7} in v[0..7],  K = {16..23} in v[8..15]
//   lanes 16-31: row = lane-16, K = {8..15},            K = {24..31}
__device__ __forceinline__ v16bf load_frag_nt(const ushort_t* p0, int stride) {
    const int lane = threadIdx.x & 31;
    const int row  = lane & 15;
    const int half = lane >> 4;
    const ushort_t* p = p0 + (size_t)row * stride + half * 8;
    v8bf lo = *(const v8bf*)(p);
    v8bf hi = *(const v8bf*)(p + 16);
    v16bf r;
#pragma unroll
    for (int i = 0; i < 8; ++i) { r[i] = lo[i]; r[i + 8] = hi[i]; }
    return r;
}

__device__ __forceinline__ v8f wmma_bf16(v16bf a, v16bf b, v8f c) {
    return __builtin_amdgcn_wmma_f32_16x16x32_bf16(
        false, a, false, b, (short)0, c, false, false);
}

#if HAVE_TDM
__device__ __forceinline__ unsigned lds_off32(const void* p) {
    // generic pointer to LDS: addr[31:0] is the LDS byte offset (ISA 10.2)
    return (unsigned)(unsigned long long)p;
}

// Issue one TDM 2D tile load: tile (tile_d1 rows x tile_d0 elems) of a
// row-major bf16 tensor (tensor_d1 rows x tensor_d0 elems, row stride
// stride0 elems) starting at gptr, packed contiguously into LDS at lds_off.
// D# layout per CDNA5 ISA 8.3/8.4 (group0: count/lds/global/type=2,
// group1: data_size=2B, dims, tile dims, dim0 stride).
__device__ __forceinline__ void tdm_load_2d(unsigned lds_off, const ushort_t* gptr,
                                            unsigned tensor_d0, unsigned tensor_d1,
                                            unsigned tile_d0, unsigned tile_d1,
                                            unsigned stride0) {
    unsigned long long ga = (unsigned long long)(size_t)gptr;
    u32x4 g0;
    g0[0] = 1u;                                                  // count=1 (valid user D#)
    g0[1] = lds_off;                                             // lds_addr
    g0[2] = (unsigned)ga;                                        // global_addr[31:0]
    g0[3] = ((unsigned)(ga >> 32) & 0x01FFFFFFu) | (2u << 30);   // ga[56:32] | type=2
    i32x8 g1;
    g1[0] = (int)(1u << 16);                                     // data_size=1 (2 bytes)
    g1[1] = (int)(tensor_d0 << 16);                              // tensor_dim0 lo16
    g1[2] = (int)((tensor_d0 >> 16) | (tensor_d1 << 16));        // td0 hi16 | td1 lo16
    g1[3] = (int)((tensor_d1 >> 16) | (tile_d0 << 16));          // td1 hi16 | tile_dim0
    g1[4] = (int)tile_d1;                                        // tile_dim1 (tile_dim2=0)
    g1[5] = (int)stride0;                                        // tensor_dim0_stride lo32
    g1[6] = 0;                                                   // stride0 hi16 | dim1_stride lo
    g1[7] = 0;
    i32x4 z4 = {0, 0, 0, 0};
#if __has_include(<hip/amd_detail/amd_gfx1250_TDM.h>)
    i32x8 z8 = {0, 0, 0, 0, 0, 0, 0, 0};                         // 6-arg toolchain
    __builtin_amdgcn_tensor_load_to_lds(g0, g1, z4, z4, z8, 0);
#else
    __builtin_amdgcn_tensor_load_to_lds(g0, g1, z4, z4, 0);      // 5-arg toolchain
#endif
}
#endif  // HAVE_TDM

// ---------------- fp32 -> bf16 conversion ----------------

__global__ void cvt_f32_bf16_kernel(const float* __restrict__ src,
                                    ushort_t* __restrict__ dst, int n) {
    int i = blockIdx.x * blockDim.x + threadIdx.x;
    if (i < n) dst[i] = f2bf(src[i]);
}

// ---------------- NT GEMM: C[m,n] = sum_k A[m,k] * W[n,k] ----------------
// A: [M,K] bf16 row-major, W: [N,K] bf16 row-major (torch Linear weight).
// mode 0: bf16 row-major C ; mode 1: bf16 scatter to Vt[b,h,dk,S] ; mode 2: fp32 C
#define BM 128
#define BN 128
#define BK 32

// Preload all fragments into distinct registers first so the LDS loads can
// stay in flight (partial s_wait_dscnt) while the 8 WMMAs issue.
__device__ __forceinline__ void mma_step(const ushort_t* As, const ushort_t* Bs,
                                         int wrow, v8f acc[8]) {
    v16bf af = load_frag_nt(&As[wrow * BK], BK);
    v16bf bfrag[8];
#pragma unroll
    for (int t = 0; t < 8; ++t) bfrag[t] = load_frag_nt(&Bs[(t * 16) * BK], BK);
#pragma unroll
    for (int t = 0; t < 8; ++t) acc[t] = wmma_bf16(af, bfrag[t], acc[t]);
}

__global__ __launch_bounds__(256)
void gemm_nt_bf16_kernel(const ushort_t* __restrict__ A,
                         const ushort_t* __restrict__ W,
                         void* __restrict__ outp,
                         int M, int N, int K, int mode) {
    __shared__ __align__(16) ushort_t As[2][BM * BK];   // 2 x 8 KB
    __shared__ __align__(16) ushort_t Bs[2][BN * BK];   // 2 x 8 KB

    const int tid   = threadIdx.x;
    const int wave  = tid >> 5;
    const int lane  = tid & 31;
    const int mbase = blockIdx.x * BM;
    const int nbase = blockIdx.y * BN;
    const int wrow  = wave * 16;

    v8f acc[8];
#pragma unroll
    for (int t = 0; t < 8; ++t) acc[t] = vzero8();

#if HAVE_TDM
    // --- TDM double-buffered pipeline: DMA tile k+1 while computing tile k ---
    const bool issuer = (wave == 0);
    if (issuer) {
        tdm_load_2d(lds_off32(&As[0][0]), &A[(size_t)mbase * K], K, M, BK, BM, K);
        tdm_load_2d(lds_off32(&Bs[0][0]), &W[(size_t)nbase * K], K, N, BK, BN, K);
        __builtin_amdgcn_s_wait_tensorcnt(0);
    }
    __syncthreads();
    for (int k0 = 0; k0 < K; k0 += BK) {
        const int cur = (k0 >> 5) & 1;
        if (issuer && (k0 + BK) < K) {
            tdm_load_2d(lds_off32(&As[cur ^ 1][0]),
                        &A[(size_t)mbase * K + (k0 + BK)], K, M, BK, BM, K);
            tdm_load_2d(lds_off32(&Bs[cur ^ 1][0]),
                        &W[(size_t)nbase * K + (k0 + BK)], K, N, BK, BN, K);
        }
        mma_step(&As[cur][0], &Bs[cur][0], wrow, acc);
        if (issuer && (k0 + BK) < K) __builtin_amdgcn_s_wait_tensorcnt(0);
        __syncthreads();
    }
#else
    // --- fallback: cooperative synchronous tile loads, single buffer ---
    for (int k0 = 0; k0 < K; k0 += BK) {
        {
            int c0 = tid, c1 = tid + 256;
            int r0 = c0 >> 2, o0 = (c0 & 3) * 8;
            int r1 = c1 >> 2, o1 = (c1 & 3) * 8;
            *(uint4*)&As[0][r0 * BK + o0] = *(const uint4*)&A[(size_t)(mbase + r0) * K + k0 + o0];
            *(uint4*)&As[0][r1 * BK + o1] = *(const uint4*)&A[(size_t)(mbase + r1) * K + k0 + o1];
            *(uint4*)&Bs[0][r0 * BK + o0] = *(const uint4*)&W[(size_t)(nbase + r0) * K + k0 + o0];
            *(uint4*)&Bs[0][r1 * BK + o1] = *(const uint4*)&W[(size_t)(nbase + r1) * K + k0 + o1];
        }
        __syncthreads();
        mma_step(&As[0][0], &Bs[0][0], wrow, acc);
        __syncthreads();
    }
#endif

    // C layout: VGPR r -> (row r, lanes 0-15 n=lane) / (row r+8, lanes 16-31)
    const int half = lane >> 4, nn = lane & 15;
#pragma unroll
    for (int t = 0; t < 8; ++t) {
#pragma unroll
        for (int r = 0; r < 8; ++r) {
            int m = mbase + wrow + r + half * 8;
            int n = nbase + t * 16 + nn;
            float v = acc[t][r];
            if (mode == 2) {
                ((float*)outp)[(size_t)m * N + n] = v;
            } else if (mode == 0) {
                ((ushort_t*)outp)[(size_t)m * N + n] = f2bf(v);
            } else {
                int b = m >> 11, s = m & (S_ - 1);
                int hh = n >> 6, dkc = n & (DK_ - 1);
                ((ushort_t*)outp)[((size_t)((b * H_ + hh) * DK_ + dkc)) * S_ + s] = f2bf(v);
            }
        }
    }
}

// ---------------- fused flash attention ----------------
// Q,K: bf16 [B,S,D] row-major (head h at col h*64). Vt: bf16 [B,H,DK,S].
// Out: bf16 [B,S,D]. Grid: (S/128, B*H), block 256 (8 waves x 16 q-rows).
__global__ __launch_bounds__(256)
void attention_kernel(const ushort_t* __restrict__ Qb,
                      const ushort_t* __restrict__ Kb,
                      const ushort_t* __restrict__ Vt,
                      ushort_t* __restrict__ Ao) {
    __shared__ __align__(16) ushort_t Pl[8][16 * 32];   // per-wave P scratch, 8 KB

    const int wave  = threadIdx.x >> 5;
    const int lane  = threadIdx.x & 31;
    const int half  = lane >> 4;
    const int nn    = lane & 15;
    const int b     = blockIdx.y >> 4;
    const int h     = blockIdx.y & (H_ - 1);
    const int qrow0 = blockIdx.x * 128 + wave * 16;
    const float kScale = 0.125f;   // 1/sqrt(64)

    const ushort_t* qbase = Qb + ((size_t)(b * S_ + qrow0) * D_ + h * DK_);
    v16bf qf0 = load_frag_nt(qbase, D_);        // dk 0..31
    v16bf qf1 = load_frag_nt(qbase + 32, D_);   // dk 32..63

    v8f o[4] = {vzero8(), vzero8(), vzero8(), vzero8()};
    float m_st[8], l_st[8];
#pragma unroll
    for (int r = 0; r < 8; ++r) { m_st[r] = -INFINITY; l_st[r] = 0.f; }

    for (int kv = 0; kv < S_; kv += 32) {
        // ---- scores: S = Q @ K^T, two 16x16 tiles along kv ----
        // preload all 4 K fragments so the global loads overlap
        const ushort_t* kb0 = Kb + ((size_t)(b * S_ + kv) * D_ + h * DK_);
        const ushort_t* kb1 = Kb + ((size_t)(b * S_ + kv + 16) * D_ + h * DK_);
        v16bf k00 = load_frag_nt(kb0, D_);
        v16bf k01 = load_frag_nt(kb0 + 32, D_);
        v16bf k10 = load_frag_nt(kb1, D_);
        v16bf k11 = load_frag_nt(kb1 + 32, D_);
        v8f s0 = vzero8(), s1 = vzero8();
        s0 = wmma_bf16(qf0, k00, s0);
        s0 = wmma_bf16(qf1, k01, s0);
        s1 = wmma_bf16(qf0, k10, s1);
        s1 = wmma_bf16(qf1, k11, s1);

        // ---- online softmax (row stats per 16-lane half) ----
        float p0[8], p1[8], alpha[8];
#pragma unroll
        for (int r = 0; r < 8; ++r) {
            float a0 = s0[r] * kScale, a1 = s1[r] * kScale;
            float mx = fmaxf(a0, a1);
            mx = fmaxf(mx, __shfl_xor(mx, 1));
            mx = fmaxf(mx, __shfl_xor(mx, 2));
            mx = fmaxf(mx, __shfl_xor(mx, 4));
            mx = fmaxf(mx, __shfl_xor(mx, 8));
            float mnew = fmaxf(m_st[r], mx);
            float al = __expf(m_st[r] - mnew);
            p0[r] = __expf(a0 - mnew);
            p1[r] = __expf(a1 - mnew);
            float ps = p0[r] + p1[r];
            ps += __shfl_xor(ps, 1);
            ps += __shfl_xor(ps, 2);
            ps += __shfl_xor(ps, 4);
            ps += __shfl_xor(ps, 8);
            l_st[r] = l_st[r] * al + ps;
            m_st[r] = mnew;
            alpha[r] = al;
        }
#pragma unroll
        for (int d = 0; d < 4; ++d)
#pragma unroll
            for (int r = 0; r < 8; ++r) o[d][r] *= alpha[r];

        // ---- P: score layout -> A-fragment layout via per-wave LDS ----
        {
            ushort_t* pl = &Pl[wave][0];
#pragma unroll
            for (int r = 0; r < 8; ++r) {
                int row = r + half * 8;
                pl[row * 32 + nn]      = f2bf(p0[r]);
                pl[row * 32 + 16 + nn] = f2bf(p1[r]);
            }
        }
        asm volatile("" ::: "memory");   // keep LDS store->load order; DS in-order per wave
        v16bf pa = load_frag_nt(&Pl[wave][0], 32);

        // ---- O += P @ V, V fragments from per-head transposed Vt ----
        v16bf vf[4];
#pragma unroll
        for (int d = 0; d < 4; ++d) {
            const ushort_t* vb = Vt + ((size_t)((b * H_ + h) * DK_ + d * 16)) * S_ + kv;
            vf[d] = load_frag_nt(vb, S_);
        }
#pragma unroll
        for (int d = 0; d < 4; ++d) o[d] = wmma_bf16(pa, vf[d], o[d]);
    }

    // ---- epilogue: normalize and store bf16 [B,S,D] ----
#pragma unroll
    for (int r = 0; r < 8; ++r) {
        float inv = 1.0f / l_st[r];
        int row = qrow0 + r + half * 8;
#pragma unroll
        for (int d = 0; d < 4; ++d) {
            int col = h * DK_ + d * 16 + nn;
            Ao[(size_t)(b * S_ + row) * D_ + col] = f2bf(o[d][r] * inv);
        }
    }
}

// ---------------- host launch ----------------

extern "C" void kernel_launch(void* const* d_in, const int* in_sizes, int n_in,
                              void* d_out, int out_size, void* d_ws, size_t ws_size,
                              hipStream_t stream) {
    const float* x  = (const float*)d_in[0];
    const float* wq = (const float*)d_in[1];
    const float* wk = (const float*)d_in[2];
    const float* wv = (const float*)d_in[3];
    const float* wo = (const float*)d_in[4];
    float* out = (float*)d_out;

    const int M = B_ * S_;            // 8192
    const int N = D_;                 // 1024
    const int K = D_;                 // 1024
    const int nX = M * D_;            // 8388608
    const int nW = D_ * D_;           // 1048576

    char* ws = (char*)d_ws;
    size_t off = 0;
    ushort_t* xb  = (ushort_t*)(ws + off); off += (size_t)nX * 2;
    ushort_t* wqb = (ushort_t*)(ws + off); off += (size_t)nW * 2;
    ushort_t* wkb = (ushort_t*)(ws + off); off += (size_t)nW * 2;
    ushort_t* wvb = (ushort_t*)(ws + off); off += (size_t)nW * 2;
    ushort_t* wob = (ushort_t*)(ws + off); off += (size_t)nW * 2;
    ushort_t* qb  = (ushort_t*)(ws + off); off += (size_t)nX * 2;
    ushort_t* kb  = (ushort_t*)(ws + off); off += (size_t)nX * 2;
    ushort_t* vtb = (ushort_t*)(ws + off); off += (size_t)nX * 2;  // transposed V
    ushort_t* ab  = (ushort_t*)(ws + off); off += (size_t)nX * 2;  // attn out

    cvt_f32_bf16_kernel<<<(nX + 255) / 256, 256, 0, stream>>>(x,  xb,  nX);
    cvt_f32_bf16_kernel<<<(nW + 255) / 256, 256, 0, stream>>>(wq, wqb, nW);
    cvt_f32_bf16_kernel<<<(nW + 255) / 256, 256, 0, stream>>>(wk, wkb, nW);
    cvt_f32_bf16_kernel<<<(nW + 255) / 256, 256, 0, stream>>>(wv, wvb, nW);
    cvt_f32_bf16_kernel<<<(nW + 255) / 256, 256, 0, stream>>>(wo, wob, nW);

    dim3 ggrid(M / BM, N / BN);
    gemm_nt_bf16_kernel<<<ggrid, 256, 0, stream>>>(xb, wqb, qb,  M, N, K, 0);
    gemm_nt_bf16_kernel<<<ggrid, 256, 0, stream>>>(xb, wkb, kb,  M, N, K, 0);
    gemm_nt_bf16_kernel<<<ggrid, 256, 0, stream>>>(xb, wvb, vtb, M, N, K, 1);

    dim3 agrid(S_ / 128, B_ * H_);
    attention_kernel<<<agrid, 256, 0, stream>>>(qb, kb, vtb, ab);

    gemm_nt_bf16_kernel<<<ggrid, 256, 0, stream>>>(ab, wob, out, M, N, K, 2);

    (void)in_sizes; (void)n_in; (void)out_size; (void)ws_size;
}
